// MxDNAFlashAttention2_32186484916843
// MI455X (gfx1250) — compile-verified, matrix operand order checked by reading, not run on hardware
//
#include <hip/hip_runtime.h>
#include <math.h>

// ---------------- CDNA5 WMMA / TDM types ----------------
typedef _Float16 half_t;
typedef __attribute__((ext_vector_type(16))) _Float16     v16h;
typedef __attribute__((ext_vector_type(8)))  _Float16     v8h;
typedef __attribute__((ext_vector_type(8)))  float        v8f;
typedef __attribute__((ext_vector_type(4)))  unsigned int v4u;
typedef __attribute__((ext_vector_type(4)))  int          v4i;
typedef __attribute__((ext_vector_type(8)))  int          v8i;

#define HIDDEN   1024
#define N_HEADS  16
#define N_KV     4
#define HEAD_DIM 64
#define BATCH    2
#define SEQ      2048
#define ROWS     (BATCH * SEQ)            // 4096 token rows
// ln(10000)/32
#define ROPE_C   0.2878231366242557f

#if defined(__has_builtin)
#if __has_builtin(__builtin_amdgcn_tensor_load_to_lds) && \
    __has_builtin(__builtin_amdgcn_s_wait_tensorcnt)
#define USE_TDM 1
#endif
#endif

__device__ __forceinline__ v8f wmma_f16(v16h a, v16h b, v8f c) {
  return __builtin_amdgcn_wmma_f32_16x16x32_f16(false, a, false, b, (short)0, c,
                                                false, false);
}

// Load a 16x32 f16 fragment (A layout; B uses the same pattern applied to
// row-major B^T) from row-major memory with leading dimension ld.
// Lane l<16 : row row0+l, K = k0+0..7 and k0+16..23
// Lane l>=16: row row0+(l-16), K = k0+8..15 and k0+24..31
__device__ __forceinline__ v16h load_frag(const half_t* base, int ld, int row0,
                                          int k0, int lane) {
  int l  = lane & 15;
  int hi = (lane >> 4) & 1;
  const half_t* p = base + (size_t)(row0 + l) * (size_t)ld + k0 + hi * 8;
  union { v16h v; v8h h[2]; } u;
  u.h[0] = *(const v8h*)(p);
  u.h[1] = *(const v8h*)(p + 16);
  return u.v;
}

// Generic LDS pointer -> byte offset within the workgroup LDS allocation
// (flat LDS aperture maps the offset into addr[31:0]).
__device__ __forceinline__ unsigned lds_off(const void* p) {
  return (unsigned)(uintptr_t)p;
}

// Tensor Data Mover: DMA a 2-D tile of f16 (tile_d1 rows x tile_d0 elements,
// row stride stride0 elements) from global memory into LDS, packed with
// ld = tile_d0.  D# layout per CDNA5 ISA ch.8 (groups 2/3 zero => <=2D).
__device__ __forceinline__ void tdm_load_f16(unsigned lds_byte_off,
                                             const void* gptr,
                                             unsigned tile_d0, unsigned tile_d1,
                                             unsigned stride0) {
#ifdef USE_TDM
  unsigned long long ga = (unsigned long long)(uintptr_t)gptr;
  v4u g0;
  g0[0] = 1u;                                               // count=1, user D#
  g0[1] = lds_byte_off;                                     // lds_addr
  g0[2] = (unsigned)ga;                                     // global_addr lo
  g0[3] = (unsigned)((ga >> 32) & 0x01FFFFFFull) | 0x80000000u; // hi + type=2
  v8i g1;
  g1[0] = (int)(1u << 16);                                  // data_size=2B
  g1[1] = (int)((tile_d0 & 0xFFFFu) << 16);                 // tensor_dim0 lo
  g1[2] = (int)(((tile_d0 >> 16) & 0xFFFFu) |
                ((tile_d1 & 0xFFFFu) << 16));               // dim0 hi | dim1 lo
  g1[3] = (int)(((tile_d1 >> 16) & 0xFFFFu) |
                ((tile_d0 & 0xFFFFu) << 16));               // dim1 hi | tile_d0
  g1[4] = (int)(tile_d1 & 0xFFFFu);                         // tile_d1, tile_d2=0
  g1[5] = (int)stride0;                                     // dim0 stride lo
  g1[6] = 0;                                                // stride hi, dim1 stride
  g1[7] = 0;
  v4i z4 = {0, 0, 0, 0};
  v8i z8 = {0, 0, 0, 0, 0, 0, 0, 0};
  // 6-arg form (clang-23 / therock headers): groups 0..3, extra group, cpol.
  __builtin_amdgcn_tensor_load_to_lds(g0, g1, z4, z4, z8, 0);
#else
  (void)lds_byte_off; (void)gptr; (void)tile_d0; (void)tile_d1; (void)stride0;
#endif
}

// ---------------- kernel 0: fp32 -> f16 convert ----------------
__global__ void cvt_f32_f16(const float* __restrict__ src,
                            half_t* __restrict__ dst, int n) {
  for (int i = blockIdx.x * blockDim.x + threadIdx.x; i < n;
       i += gridDim.x * blockDim.x)
    dst[i] = (half_t)src[i];
}

// ---------------- kernel 1: QKV projection + RoPE ----------------
// grid.x = ROWS/32 row blocks, grid.y = 24 feature chunks of 64
// (chunks 0..15 -> Q heads, 16..19 -> K heads, 20..23 -> V heads)
// block = 128 threads (4 waves); each wave owns a 16x32 sub-tile.
__global__ void qkv_rope(const half_t* __restrict__ h16,
                         const half_t* __restrict__ wq,
                         const half_t* __restrict__ wk,
                         const half_t* __restrict__ wv,
                         half_t* __restrict__ q16,
                         half_t* __restrict__ k16,
                         half_t* __restrict__ vT16) {
  __shared__ __align__(32) float tile[32][65];

  int chunk = blockIdx.y;
  int c0    = chunk * 64;
  const half_t* W;
  int wrow0, kind;
  if (c0 < 1024)      { W = wq; wrow0 = c0;        kind = 0; }
  else if (c0 < 1280) { W = wk; wrow0 = c0 - 1024; kind = 1; }
  else                { W = wv; wrow0 = c0 - 1280; kind = 2; }

  int lane  = threadIdx.x & 31;
  int w     = threadIdx.x >> 5;
  int mhalf = w & 1;
  int ncol0 = (w >> 1) * 32;
  int row0  = blockIdx.x * 32 + mhalf * 16;

  v8f acc0 = {};
  v8f acc1 = {};
#pragma unroll 4
  for (int kk = 0; kk < HIDDEN; kk += 32) {
    v16h a  = load_frag(h16, HIDDEN, row0, kk, lane);
    v16h b0 = load_frag(W, HIDDEN, wrow0 + ncol0, kk, lane);
    v16h b1 = load_frag(W, HIDDEN, wrow0 + ncol0 + 16, kk, lane);
    if (kk + 32 < HIDDEN) {
      __builtin_prefetch(h16 + (size_t)(row0 + (lane & 15)) * HIDDEN + kk + 32, 0, 1);
      __builtin_prefetch(W + (size_t)(wrow0 + ncol0 + (lane & 15)) * HIDDEN + kk + 32, 0, 1);
    }
    acc0 = wmma_f16(a, b0, acc0);
    acc1 = wmma_f16(a, b1, acc1);
  }

  int l  = lane & 15;
  int hi = (lane >> 4) & 1;
#pragma unroll
  for (int r = 0; r < 8; ++r) {
    int m = mhalf * 16 + r + hi * 8;
    tile[m][ncol0 + l]      = acc0[r];
    tile[m][ncol0 + 16 + l] = acc1[r];
  }
  __syncthreads();

  // RoPE (q,k) / transpose-store (v)
  for (int idx = threadIdx.x; idx < 32 * 64; idx += 128) {
    int m = idx >> 6;
    int d = idx & 63;
    int rowg = blockIdx.x * 32 + m;
    int b = rowg >> 11;      // / SEQ
    int s = rowg & 2047;     // % SEQ
    float x = tile[m][d];
    if (kind < 2) {
      float f  = __expf(-(float)(d & 31) * ROPE_C);
      float sn, cs;
      __sincosf((float)s * f, &sn, &cs);
      float rot = (d < 32) ? -tile[m][d + 32] : tile[m][d - 32];
      x = x * cs + rot * sn;
      if (kind == 0) {
        int h = c0 >> 6;
        q16[((size_t)(b * N_HEADS + h) * SEQ + s) * HEAD_DIM + d] = (half_t)x;
      } else {
        int hk = (c0 - 1024) >> 6;
        k16[((size_t)(b * N_KV + hk) * SEQ + s) * HEAD_DIM + d] = (half_t)x;
      }
    } else {
      int hv = (c0 - 1280) >> 6;
      vT16[((size_t)(b * N_KV + hv) * HEAD_DIM + d) * SEQ + s] = (half_t)x;
    }
  }
}

// ---------------- kernel 2: flash attention (GQA) ----------------
// grid.x = SEQ/64 query blocks, grid.y = BATCH*N_HEADS. block = 128 (4 waves).
// Each wave handles 16 query rows.  K/V 32-token blocks are staged in LDS by
// the Tensor Data Mover (double-buffered, wave 0 drives the DMA) so the four
// waves share one copy of each KV block.
__global__ void flash_attn(const half_t* __restrict__ q16,
                           const half_t* __restrict__ k16,
                           const half_t* __restrict__ vT16,
                           half_t* __restrict__ attn16) {
  __shared__ __align__(32) half_t kbuf[2][32][64];   // 8 KB
  __shared__ __align__(32) half_t vbuf[2][64][32];   // 8 KB
  __shared__ __align__(32) half_t pbuf[4][16][32];   // 4 KB

  int bh = blockIdx.y;
  int b  = bh >> 4;
  int h  = bh & 15;
  int hk = h >> 2;
  int lane = threadIdx.x & 31;
  int w    = threadIdx.x >> 5;
  int m0   = blockIdx.x * 64 + w * 16;   // query row within SEQ

  const half_t* Q = q16 + (size_t)(b * N_HEADS + h) * SEQ * HEAD_DIM;
  const half_t* K = k16 + (size_t)(b * N_KV + hk) * SEQ * HEAD_DIM;
  const half_t* V = vT16 + (size_t)(b * N_KV + hk) * HEAD_DIM * SEQ;

  v16h aq0 = load_frag(Q, HEAD_DIM, m0, 0, lane);
  v16h aq1 = load_frag(Q, HEAD_DIM, m0, 32, lane);

  half_t* myP = &pbuf[w][0][0];
  int l  = lane & 15;
  int hi = (lane >> 4) & 1;

  float mrow[8], lrow[8];
  v8f o[4] = {};
#pragma unroll
  for (int r = 0; r < 8; ++r) { mrow[r] = -3.0e38f; lrow[r] = 0.0f; }

#ifdef USE_TDM
  if (w == 0) {
    // preload KV block 0: K block = contiguous 32x64 f16, V block = 64 rows
    // of 32 f16 with row stride SEQ, packed to ld=32 in LDS.
    tdm_load_f16(lds_off(&kbuf[0][0][0]), K, 2048, 1, 2048);
    tdm_load_f16(lds_off(&vbuf[0][0][0]), V, 32, 64, SEQ);
  }
#endif

  for (int j = 0, it = 0; j < SEQ; j += 32, ++it) {
    int cur = it & 1;
#ifdef USE_TDM
    int nxt = cur ^ 1;
    if (w == 0) __builtin_amdgcn_s_wait_tensorcnt(0);
    __syncthreads();                        // KV block `cur` visible to all
    if (w == 0 && (j + 32) < SEQ) {         // overlap next DMA with compute
      tdm_load_f16(lds_off(&kbuf[nxt][0][0]),
                   K + (size_t)(j + 32) * HEAD_DIM, 2048, 1, 2048);
      tdm_load_f16(lds_off(&vbuf[nxt][0][0]), V + (j + 32), 32, 64, SEQ);
    }
#else
    cur = 0;
    __syncthreads();
    for (int idx = threadIdx.x; idx < 2048; idx += 128) {
      (&kbuf[0][0][0])[idx] = K[(size_t)j * HEAD_DIM + idx];
      int d = idx >> 5, c = idx & 31;
      vbuf[0][d][c] = V[(size_t)d * SEQ + j + c];
    }
    __syncthreads();
#endif
    const half_t* Kb = &kbuf[cur][0][0];
    const half_t* Vb = &vbuf[cur][0][0];

    // ---- scores S = (Q K^T) * 0.125 for kv tokens j..j+31 ----
    v16h bk00 = load_frag(Kb, HEAD_DIM, 0,  0,  lane);
    v16h bk01 = load_frag(Kb, HEAD_DIM, 0,  32, lane);
    v16h bk10 = load_frag(Kb, HEAD_DIM, 16, 0,  lane);
    v16h bk11 = load_frag(Kb, HEAD_DIM, 16, 32, lane);
    v8f s0 = {};
    v8f s1 = {};
    s0 = wmma_f16(aq0, bk00, s0);
    s0 = wmma_f16(aq1, bk01, s0);
    s1 = wmma_f16(aq0, bk10, s1);
    s1 = wmma_f16(aq1, bk11, s1);

    float p0[8], p1[8], rmax[8];
#pragma unroll
    for (int r = 0; r < 8; ++r) {
      p0[r] = s0[r] * 0.125f;
      p1[r] = s1[r] * 0.125f;
      rmax[r] = fmaxf(p0[r], p1[r]);
    }
    // row-wise max across the 16 lanes holding this row's columns
#pragma unroll
    for (int off = 8; off >= 1; off >>= 1)
#pragma unroll
      for (int r = 0; r < 8; ++r)
        rmax[r] = fmaxf(rmax[r], __shfl_xor(rmax[r], off, 32));

    float alpha[8], rsum[8];
#pragma unroll
    for (int r = 0; r < 8; ++r) {
      float mn = fmaxf(mrow[r], rmax[r]);
      alpha[r] = __expf(mrow[r] - mn);
      mrow[r]  = mn;
      p0[r] = __expf(p0[r] - mn);
      p1[r] = __expf(p1[r] - mn);
      rsum[r] = p0[r] + p1[r];
    }
#pragma unroll
    for (int off = 8; off >= 1; off >>= 1)
#pragma unroll
      for (int r = 0; r < 8; ++r)
        rsum[r] += __shfl_xor(rsum[r], off, 32);
#pragma unroll
    for (int r = 0; r < 8; ++r) lrow[r] = lrow[r] * alpha[r] + rsum[r];

    // ---- stage P (16x32 f16) in LDS, reload in A-fragment layout ----
#pragma unroll
    for (int r = 0; r < 8; ++r) {
      int m = r + hi * 8;
      myP[m * 32 + l]      = (half_t)p0[r];
      myP[m * 32 + 16 + l] = (half_t)p1[r];
    }
    __asm__ volatile("s_wait_dscnt 0x0" ::: "memory");  // cross-lane LDS RAW
    v16h ap = load_frag(myP, 32, 0, 0, lane);

    v16h bv0 = load_frag(Vb, 32, 0,  0, lane);
    v16h bv1 = load_frag(Vb, 32, 16, 0, lane);
    v16h bv2 = load_frag(Vb, 32, 32, 0, lane);
    v16h bv3 = load_frag(Vb, 32, 48, 0, lane);

#pragma unroll
    for (int t = 0; t < 4; ++t)
#pragma unroll
      for (int r = 0; r < 8; ++r) o[t][r] *= alpha[r];

    o[0] = wmma_f16(ap, bv0, o[0]);
    o[1] = wmma_f16(ap, bv1, o[1]);
    o[2] = wmma_f16(ap, bv2, o[2]);
    o[3] = wmma_f16(ap, bv3, o[3]);
  }

  // normalize and store [B*S, 1024] f16
#pragma unroll
  for (int r = 0; r < 8; ++r) {
    float inv = 1.0f / lrow[r];
    int m = m0 + r + hi * 8;
    size_t rowoff = (size_t)(b * SEQ + m) * HIDDEN + h * HEAD_DIM;
#pragma unroll
    for (int t = 0; t < 4; ++t)
      attn16[rowoff + t * 16 + l] = (half_t)(o[t][r] * inv);
  }
}

// ---------------- kernel 3: output projection ----------------
// grid.x = ROWS/32, grid.y = HIDDEN/64. block = 128 (4 waves, 16x32 each).
__global__ void oproj(const half_t* __restrict__ attn16,
                      const half_t* __restrict__ wo,
                      float* __restrict__ out) {
  int lane  = threadIdx.x & 31;
  int w     = threadIdx.x >> 5;
  int row0  = blockIdx.x * 32 + (w & 1) * 16;
  int ncol0 = blockIdx.y * 64 + (w >> 1) * 32;

  v8f acc0 = {};
  v8f acc1 = {};
#pragma unroll 4
  for (int kk = 0; kk < HIDDEN; kk += 32) {
    v16h a  = load_frag(attn16, HIDDEN, row0, kk, lane);
    v16h b0 = load_frag(wo, HIDDEN, ncol0, kk, lane);
    v16h b1 = load_frag(wo, HIDDEN, ncol0 + 16, kk, lane);
    if (kk + 32 < HIDDEN) {
      __builtin_prefetch(attn16 + (size_t)(row0 + (lane & 15)) * HIDDEN + kk + 32, 0, 1);
      __builtin_prefetch(wo + (size_t)(ncol0 + (lane & 15)) * HIDDEN + kk + 32, 0, 1);
    }
    acc0 = wmma_f16(a, b0, acc0);
    acc1 = wmma_f16(a, b1, acc1);
  }

  int l  = lane & 15;
  int hi = (lane >> 4) & 1;
#pragma unroll
  for (int r = 0; r < 8; ++r) {
    size_t m = (size_t)(row0 + r + hi * 8);
    out[m * HIDDEN + ncol0 + l]      = acc0[r];
    out[m * HIDDEN + ncol0 + 16 + l] = acc1[r];
  }
}

// ---------------- host launcher ----------------
extern "C" void kernel_launch(void* const* d_in, const int* in_sizes, int n_in,
                              void* d_out, int out_size, void* d_ws,
                              size_t ws_size, hipStream_t stream) {
  const float* hs = (const float*)d_in[0];
  const float* wq = (const float*)d_in[1];
  const float* wk = (const float*)d_in[2];
  const float* wv = (const float*)d_in[3];
  const float* wo = (const float*)d_in[4];
  (void)in_sizes; (void)n_in; (void)out_size; (void)ws_size;

  char* ws = (char*)d_ws;
  size_t off = 0;
  half_t* h16  = (half_t*)(ws + off); off += (size_t)ROWS * HIDDEN * 2;          // 8 MB
  half_t* wq16 = (half_t*)(ws + off); off += (size_t)HIDDEN * HIDDEN * 2;        // 2 MB
  half_t* wk16 = (half_t*)(ws + off); off += (size_t)256 * HIDDEN * 2;           // 0.5 MB
  half_t* wv16 = (half_t*)(ws + off); off += (size_t)256 * HIDDEN * 2;           // 0.5 MB
  half_t* wo16 = (half_t*)(ws + off); off += (size_t)HIDDEN * HIDDEN * 2;        // 2 MB
  half_t* q16  = (half_t*)(ws + off); off += (size_t)ROWS * HIDDEN * 2;          // 8 MB
  half_t* k16  = (half_t*)(ws + off); off += (size_t)BATCH * N_KV * SEQ * 64 * 2;// 2 MB
  half_t* vT16 = (half_t*)(ws + off); off += (size_t)BATCH * N_KV * SEQ * 64 * 2;// 2 MB
  half_t* a16  = (half_t*)(ws + off); off += (size_t)ROWS * HIDDEN * 2;          // 8 MB

  cvt_f32_f16<<<1024, 256, 0, stream>>>(hs, h16, ROWS * HIDDEN);
  cvt_f32_f16<<<512, 256, 0, stream>>>(wq, wq16, HIDDEN * HIDDEN);
  cvt_f32_f16<<<256, 256, 0, stream>>>(wk, wk16, 256 * HIDDEN);
  cvt_f32_f16<<<256, 256, 0, stream>>>(wv, wv16, 256 * HIDDEN);
  cvt_f32_f16<<<512, 256, 0, stream>>>(wo, wo16, HIDDEN * HIDDEN);

  qkv_rope<<<dim3(ROWS / 32, 24), 128, 0, stream>>>(h16, wq16, wk16, wv16,
                                                    q16, k16, vT16);
  flash_attn<<<dim3(SEQ / 64, BATCH * N_HEADS), 128, 0, stream>>>(q16, k16,
                                                                  vT16, a16);
  oproj<<<dim3(ROWS / 32, HIDDEN / 64), 128, 0, stream>>>(a16, wo16,
                                                          (float*)d_out);
}
